// UltimateGenomeXHybrid_33526514712901
// MI455X (gfx1250) — matrix operand order repaired and checked by reading, not compiled, not run on hardware
//
#include <hip/hip_runtime.h>
#include <hip/hip_bf16.h>
#include <math.h>

typedef __attribute__((ext_vector_type(16))) _Float16 v16h;
typedef __attribute__((ext_vector_type(8)))  _Float16 h8;
typedef __attribute__((ext_vector_type(8)))  float    v8f;

#define N_NODES 50000
#define N_EDGES 800000
#define E_TOT   (N_EDGES + N_NODES)
#define BATCH   1024
#define HID     128
#define NEG_SLOPE 0.2f
#define BN_EPS  1e-5f

// ---------------------------------------------------------------------------
// Pack fp32 [R,C] -> fp16 [R,KP], zero-padded columns (KP multiple of 32).
// ---------------------------------------------------------------------------
__global__ void pack_f16(const float* __restrict__ in, _Float16* __restrict__ out,
                         long long R, int C, int KP) {
  long long t = (long long)blockIdx.x * blockDim.x + threadIdx.x;
  if (t >= R * KP) return;
  long long r = t / KP;
  int c = (int)(t - r * KP);
  out[t] = (c < C) ? (_Float16)in[r * C + c] : (_Float16)0.f;
}

// ---------------------------------------------------------------------------
// WMMA GEMM on packed fp16: C[M,N] = act(A[M,KP] @ W[N,KP]^T + bias)
// One wave per 16x16 tile, 4 waves/block. KP compile-time, multiple of 32.
// Inner loop: 4x global_load_b128 + 1x v_wmma_f32_16x16x32_f16.
// ---------------------------------------------------------------------------
template<int KP, bool HASBIAS, bool RELU>
__global__ void gemm_wmma_h(const _Float16* __restrict__ A,
                            const _Float16* __restrict__ W,   // [N,KP] packed
                            const float* __restrict__ bias,
                            float* __restrict__ C,
                            int N, int tiles) {
  const int tile = blockIdx.x * (blockDim.x >> 5) + (threadIdx.x >> 5);
  if (tile >= tiles) return;
  const int lane = threadIdx.x & 31;
  const int ntiles = N >> 4;
  const int mt = tile / ntiles;
  const int nt = tile - mt * ntiles;
  const int hi = lane >> 4;                 // half-wave
  const int lm = lane & 15;

  // A fragment (16-bit A 16x32): lane holds M=lm; K(j) = hi*16 + j
  //   -> two contiguous 8-half chunks at k0+hi*16 and k0+hi*16+8.
  // B fragment (16-bit B 32x16): lane holds N=lm; K(j) = hi*16 + j
  //   -> one contiguous 16-half chunk at k0+hi*16.
  const _Float16* __restrict__ Arow = A + (size_t)((mt << 4) + lm) * KP + (hi << 4);
  const _Float16* __restrict__ Wrow = W + (size_t)((nt << 4) + lm) * KP + (hi << 4);

  v8f acc = {0.f, 0.f, 0.f, 0.f, 0.f, 0.f, 0.f, 0.f};
#pragma unroll
  for (int k0 = 0; k0 < KP; k0 += 32) {
    h8 a0 = *(const h8*)(Arow + k0);
    h8 a1 = *(const h8*)(Arow + k0 + 8);
    h8 b0 = *(const h8*)(Wrow + k0);
    h8 b1 = *(const h8*)(Wrow + k0 + 8);
    v16h av = __builtin_shufflevector(a0, a1, 0,1,2,3,4,5,6,7,8,9,10,11,12,13,14,15);
    v16h bv = __builtin_shufflevector(b0, b1, 0,1,2,3,4,5,6,7,8,9,10,11,12,13,14,15);
    acc = __builtin_amdgcn_wmma_f32_16x16x32_f16(false, av, false, bv,
                                                 (short)0, acc, false, false);
  }

  // C/D layout: VGPR r -> row (mt*16 + r + hi*8), col (nt*16 + lm)
  const int col   = (nt << 4) + lm;
  const int rbase = (mt << 4) + (hi << 3);
  float badd = HASBIAS ? bias[col] : 0.f;
#pragma unroll
  for (int r = 0; r < 8; ++r) {
    float v = acc[r] + badd;
    if (RELU) v = fmaxf(v, 0.f);
    C[(size_t)(rbase + r) * N + col] = v;
  }
}

// ---------------------------------------------------------------------------
// Elementwise / scatter helpers
// ---------------------------------------------------------------------------
__global__ void fill_kernel(float* __restrict__ p, float v, long long n) {
  long long t = (long long)blockIdx.x * blockDim.x + threadIdx.x;
  if (t < n) p[t] = v;
}

__global__ void node_alphas(const float* __restrict__ h,
                            const float* __restrict__ asrc,
                            const float* __restrict__ adst,
                            float* __restrict__ as_, float* __restrict__ ad_) {
  int n = blockIdx.x * blockDim.x + threadIdx.x;
  if (n >= N_NODES) return;
  const float* hr = h + (size_t)n * HID;
  float sa = 0.f, sd = 0.f;
#pragma unroll 4
  for (int k = 0; k < HID; ++k) { float v = hr[k]; sa += v * asrc[k]; sd += v * adst[k]; }
  as_[n] = sa; ad_[n] = sd;
}

__device__ __forceinline__ void atomicMaxF32(float* addr, float val) {
  if (val >= 0.f) atomicMax((int*)addr, __float_as_int(val));
  else            atomicMin((unsigned int*)addr, __float_as_uint(val));
}

__device__ __forceinline__ void edge_ends(const int* __restrict__ ei, int i, int& s, int& d) {
  if (i < N_EDGES) { s = ei[i]; d = ei[N_EDGES + i]; }
  else             { s = i - N_EDGES; d = s; }        // appended self-loops
}

__global__ void edge_logits_max(const int* __restrict__ ei,
                                const float* __restrict__ as_, const float* __restrict__ ad_,
                                float* __restrict__ e, float* __restrict__ m) {
  int i = blockIdx.x * blockDim.x + threadIdx.x;
  if (i >= E_TOT) return;
  int s, d; edge_ends(ei, i, s, d);
  float v = as_[s] + ad_[d];
  v = (v > 0.f) ? v : NEG_SLOPE * v;                  // leaky_relu
  e[i] = v;
  atomicMaxF32(&m[d], v);
}

__global__ void edge_exp_den(const int* __restrict__ ei,
                             float* __restrict__ e, const float* __restrict__ m,
                             float* __restrict__ den) {
  int i = blockIdx.x * blockDim.x + threadIdx.x;
  if (i >= E_TOT) return;
  int s, d; edge_ends(ei, i, s, d); (void)s;
  float ex = __expf(e[i] - m[d]);
  e[i] = ex;                                          // in-place: e now holds exp()
  atomicAdd(&den[d], ex);
}

__global__ void edge_scatter(const int* __restrict__ ei,
                             const float* __restrict__ ex, const float* __restrict__ den,
                             const float* __restrict__ h, float* __restrict__ agg) {
  long long t = (long long)blockIdx.x * blockDim.x + threadIdx.x;
  if (t >= (long long)E_TOT * HID) return;
  int i = (int)(t >> 7);
  int k = (int)(t & (HID - 1));
  int s, d; edge_ends(ei, i, s, d);
  float coef = ex[i] / (den[d] + 1e-16f);
  atomicAdd(&agg[(size_t)d * HID + k], coef * h[(size_t)s * HID + k]);
}

__global__ void bias_relu_inplace(float* __restrict__ x, const float* __restrict__ b,
                                  long long rows, int cols) {
  long long t = (long long)blockIdx.x * blockDim.x + threadIdx.x;
  if (t >= rows * cols) return;
  int c = (int)(t % cols);
  x[t] = fmaxf(x[t] + b[c], 0.f);
}

// ---------------------------------------------------------------------------
// Pooling
// ---------------------------------------------------------------------------
__global__ void pool_count(const int* __restrict__ batch, float* __restrict__ counts) {
  int n = blockIdx.x * blockDim.x + threadIdx.x;
  if (n >= N_NODES) return;
  atomicAdd(&counts[batch[n]], 1.f);
}
__global__ void pool_sum(const int* __restrict__ batch, const float* __restrict__ h,
                         float* __restrict__ ge) {
  long long t = (long long)blockIdx.x * blockDim.x + threadIdx.x;
  if (t >= (long long)N_NODES * HID) return;
  int n = (int)(t >> 7), k = (int)(t & (HID - 1));
  atomicAdd(&ge[(size_t)batch[n] * HID + k], h[t]);
}
__global__ void pool_div(float* __restrict__ ge, const float* __restrict__ counts) {
  int t = blockIdx.x * blockDim.x + threadIdx.x;
  if (t >= BATCH * HID) return;
  ge[t] /= fmaxf(counts[t >> 7], 1.f);
}

// ---------------------------------------------------------------------------
// LSTM gate nonlinearity (zero initial h,c; gates = x@Wih^T; biases added here)
// ---------------------------------------------------------------------------
__device__ __forceinline__ float sigm(float x) { return 1.f / (1.f + __expf(-x)); }

__global__ void lstm_gates(const float* __restrict__ g,
                           const float* __restrict__ bih, const float* __restrict__ bhh,
                           float* __restrict__ seq, int off) {
  int t = blockIdx.x * blockDim.x + threadIdx.x;
  if (t >= BATCH * 64) return;
  int b = t >> 6, j = t & 63;
  const float* gb = g + (size_t)b * 256;
  float ii = gb[j]       + bih[j]       + bhh[j];
  float gg = gb[128 + j] + bih[128 + j] + bhh[128 + j];
  float oo = gb[192 + j] + bih[192 + j] + bhh[192 + j];
  float c = sigm(ii) * tanhf(gg);
  seq[(size_t)b * HID + off + j] = sigm(oo) * tanhf(c);
}

// ---------------------------------------------------------------------------
// Tabular branch + BatchNorm (training-mode biased stats, per column)
// ---------------------------------------------------------------------------
__global__ void tab_linear(const float* __restrict__ tf, const float* __restrict__ W,
                           const float* __restrict__ b, float* __restrict__ y) {
  int t = blockIdx.x * blockDim.x + threadIdx.x;
  if (t >= BATCH * 64) return;
  int bi = t >> 6, j = t & 63;
  float s = b[j];
#pragma unroll
  for (int k = 0; k < 7; ++k) s += tf[bi * 7 + k] * W[j * 7 + k];
  y[t] = s;
}

__global__ void bn_relu_cols(const float* __restrict__ x, const float* __restrict__ g,
                             const float* __restrict__ bt, float* __restrict__ y,
                             int rows, int cols) {
  int j = blockIdx.x * blockDim.x + threadIdx.x;
  if (j >= cols) return;
  float s = 0.f, s2 = 0.f;
  for (int r = 0; r < rows; ++r) { float v = x[(size_t)r * cols + j]; s += v; s2 += v * v; }
  float mu  = s / rows;
  float var = s2 / rows - mu * mu;
  float inv = rsqrtf(var + BN_EPS) * g[j];
  for (int r = 0; r < rows; ++r) {
    float v = (x[(size_t)r * cols + j] - mu) * inv + bt[j];
    y[(size_t)r * cols + j] = fmaxf(v, 0.f);
  }
}

__global__ void concat_z(const float* __restrict__ ge, const float* __restrict__ seq,
                         const float* __restrict__ tab, float* __restrict__ z) {
  int t = blockIdx.x * blockDim.x + threadIdx.x;
  if (t >= BATCH * 320) return;
  int b = t / 320, c = t - b * 320;
  float v;
  if (c < 128)      v = ge[(size_t)b * 128 + c];
  else if (c < 256) v = seq[(size_t)b * 128 + (c - 128)];
  else              v = tab[(size_t)b * 64 + (c - 256)];
  z[t] = v;
}

__global__ void final_head(const float* __restrict__ z2, const float* __restrict__ w,
                           const float* __restrict__ b, float* __restrict__ out) {
  int bi = blockIdx.x * blockDim.x + threadIdx.x;
  if (bi >= BATCH) return;
  float s = b[0];
#pragma unroll 4
  for (int k = 0; k < 128; ++k) s += z2[(size_t)bi * 128 + k] * w[k];
  out[bi] = s;
}

// ---------------------------------------------------------------------------
// Host-side orchestration
// ---------------------------------------------------------------------------
#define LAUNCH1D(K, N, ...) K<<<(int)(((long long)(N) + 255) / 256), 256, 0, stream>>>(__VA_ARGS__)
#define PACK(src, dst, R, C, KP) \
  LAUNCH1D(pack_f16, (long long)(R) * (KP), src, dst, (long long)(R), C, KP)
#define GEMM(KP, HB, RL, Ah, Wh, bias, Cp, M, N) do {                          \
    int tiles_ = ((M) / 16) * ((N) / 16);                                      \
    gemm_wmma_h<KP, HB, RL><<<(tiles_ + 3) / 4, 128, 0, stream>>>(             \
        Ah, Wh, bias, Cp, N, tiles_);                                          \
  } while (0)

extern "C" void kernel_launch(void* const* d_in, const int* in_sizes, int n_in,
                              void* d_out, int out_size, void* d_ws, size_t ws_size,
                              hipStream_t stream) {
  const float* x      = (const float*)d_in[0];
  const int*   ei     = (const int*)  d_in[1];
  const int*   batch  = (const int*)  d_in[2];
  const float* esm    = (const float*)d_in[3];
  const float* tabf   = (const float*)d_in[4];
  const float* g1W    = (const float*)d_in[5];
  const float* g1as   = (const float*)d_in[6];
  const float* g1ad   = (const float*)d_in[7];
  const float* g1b    = (const float*)d_in[8];
  const float* g2W    = (const float*)d_in[9];
  const float* g2as   = (const float*)d_in[10];
  const float* g2ad   = (const float*)d_in[11];
  const float* g2b    = (const float*)d_in[12];
  const float* WihF   = (const float*)d_in[13];
  const float* bihF   = (const float*)d_in[15];
  const float* bhhF   = (const float*)d_in[16];
  const float* WihR   = (const float*)d_in[17];
  const float* bihR   = (const float*)d_in[19];
  const float* bhhR   = (const float*)d_in[20];
  const float* tabW   = (const float*)d_in[21];
  const float* tabB   = (const float*)d_in[22];
  const float* tbnG   = (const float*)d_in[23];
  const float* tbnB   = (const float*)d_in[24];
  const float* f1W    = (const float*)d_in[25];
  const float* f1b    = (const float*)d_in[26];
  const float* fbnG   = (const float*)d_in[27];
  const float* fbnB   = (const float*)d_in[28];
  const float* f2W    = (const float*)d_in[29];
  const float* f2b    = (const float*)d_in[30];
  const float* f3W    = (const float*)d_in[31];
  const float* f3b    = (const float*)d_in[32];
  float* out = (float*)d_out;

  // ---- workspace carve-up (float units, 16B-aligned chunks) ----
  float* ws = (float*)d_ws;
  size_t off = 0;
  auto carve = [&](size_t nfloats) {
    float* p = ws + off;
    off += (nfloats + 3) & ~(size_t)3;    // keep 16B alignment for b128 loads
    return p;
  };
  float* hbuf   = carve((size_t)N_NODES * HID);     // transformed features
  float* aggbuf = carve((size_t)N_NODES * HID);     // aggregation / layer output
  float* asv    = carve(N_NODES);
  float* adv    = carve(N_NODES);
  float* mbuf   = carve(N_NODES);
  float* denb   = carve(N_NODES);
  float* ebuf   = carve(E_TOT);
  float* ge     = carve((size_t)BATCH * HID);
  float* counts = carve(BATCH);
  float* gates  = carve((size_t)BATCH * 256);
  float* seq    = carve((size_t)BATCH * HID);
  float* tlin   = carve((size_t)BATCH * 64);
  float* tembed = carve((size_t)BATCH * 64);
  float* zbuf   = carve((size_t)BATCH * 320);
  float* z1     = carve((size_t)BATCH * 256);
  float* z1b    = carve((size_t)BATCH * 256);
  float* z2     = carve((size_t)BATCH * 128);
  _Float16* Ah  = (_Float16*)carve((size_t)N_NODES * HID / 2);  // packed activations
  _Float16* Wh  = (_Float16*)carve((size_t)256 * 480 / 2);      // packed weights
  (void)ws_size; (void)n_in; (void)in_sizes; (void)out_size;

  // ---- one GAT layer: input [N_NODES,K] (pre-packed in Ah), output -> aggbuf ----
  auto gat_attention = [&](const float* bias) {
    LAUNCH1D(node_alphas, N_NODES, hbuf, nullptr, nullptr, asv, adv); // placeholder, replaced below
  };
  (void)gat_attention;

  // ======== GAT layer 1 (K=21 -> pad 32) ========
  PACK(x,   Ah, N_NODES, 21, 32);
  PACK(g1W, Wh, HID,     21, 32);
  GEMM(32, false, false, Ah, Wh, nullptr, hbuf, N_NODES, HID);
  LAUNCH1D(node_alphas, N_NODES, hbuf, g1as, g1ad, asv, adv);
  LAUNCH1D(fill_kernel, N_NODES, mbuf, -HUGE_VALF, (long long)N_NODES);
  LAUNCH1D(fill_kernel, N_NODES, denb, 0.f, (long long)N_NODES);
  LAUNCH1D(fill_kernel, (long long)N_NODES * HID, aggbuf, 0.f, (long long)N_NODES * HID);
  LAUNCH1D(edge_logits_max, E_TOT, ei, asv, adv, ebuf, mbuf);
  LAUNCH1D(edge_exp_den, E_TOT, ei, ebuf, mbuf, denb);
  LAUNCH1D(edge_scatter, (long long)E_TOT * HID, ei, ebuf, denb, hbuf, aggbuf);
  LAUNCH1D(bias_relu_inplace, (long long)N_NODES * HID, aggbuf, g1b, (long long)N_NODES, HID);

  // ======== GAT layer 2 (K=128) ========
  PACK(aggbuf, Ah, N_NODES, HID, HID);
  PACK(g2W,    Wh, HID,     HID, HID);
  GEMM(128, false, false, Ah, Wh, nullptr, hbuf, N_NODES, HID);
  LAUNCH1D(node_alphas, N_NODES, hbuf, g2as, g2ad, asv, adv);
  LAUNCH1D(fill_kernel, N_NODES, mbuf, -HUGE_VALF, (long long)N_NODES);
  LAUNCH1D(fill_kernel, N_NODES, denb, 0.f, (long long)N_NODES);
  LAUNCH1D(fill_kernel, (long long)N_NODES * HID, aggbuf, 0.f, (long long)N_NODES * HID);
  LAUNCH1D(edge_logits_max, E_TOT, ei, asv, adv, ebuf, mbuf);
  LAUNCH1D(edge_exp_den, E_TOT, ei, ebuf, mbuf, denb);
  LAUNCH1D(edge_scatter, (long long)E_TOT * HID, ei, ebuf, denb, hbuf, aggbuf);
  LAUNCH1D(bias_relu_inplace, (long long)N_NODES * HID, aggbuf, g2b, (long long)N_NODES, HID);

  // ---- mean pooling per graph ----
  LAUNCH1D(fill_kernel, BATCH * HID, ge, 0.f, (long long)BATCH * HID);
  LAUNCH1D(fill_kernel, BATCH, counts, 0.f, (long long)BATCH);
  LAUNCH1D(pool_count, N_NODES, batch, counts);
  LAUNCH1D(pool_sum, (long long)N_NODES * HID, batch, aggbuf, ge);
  LAUNCH1D(pool_div, BATCH * HID, ge, counts);

  // ---- BiLSTM single step (K=480) ----
  PACK(esm, Ah, BATCH, 480, 480);
  PACK(WihF, Wh, 256, 480, 480);
  GEMM(480, false, false, Ah, Wh, nullptr, gates, BATCH, 256);
  LAUNCH1D(lstm_gates, BATCH * 64, gates, bihF, bhhF, seq, 0);
  PACK(WihR, Wh, 256, 480, 480);
  GEMM(480, false, false, Ah, Wh, nullptr, gates, BATCH, 256);
  LAUNCH1D(lstm_gates, BATCH * 64, gates, bihR, bhhR, seq, 64);

  // ---- tabular branch ----
  LAUNCH1D(tab_linear, BATCH * 64, tabf, tabW, tabB, tlin);
  bn_relu_cols<<<1, 64, 0, stream>>>(tlin, tbnG, tbnB, tembed, BATCH, 64);

  // ---- fusion head ----
  LAUNCH1D(concat_z, BATCH * 320, ge, seq, tembed, zbuf);
  PACK(zbuf, Ah, BATCH, 320, 320);
  PACK(f1W,  Wh, 256,   320, 320);
  GEMM(320, true, false, Ah, Wh, f1b, z1, BATCH, 256);
  bn_relu_cols<<<1, 256, 0, stream>>>(z1, fbnG, fbnB, z1b, BATCH, 256);
  PACK(z1b, Ah, BATCH, 256, 256);
  PACK(f2W, Wh, HID,   256, 256);
  GEMM(256, true, true, Ah, Wh, f2b, z2, BATCH, HID);
  LAUNCH1D(final_head, BATCH, z2, f3W, f3b, out);
}
